// LlamaFastMLP_25031069401193
// MI455X (gfx1250) — compile-verified, matrix operand order checked by reading, not run on hardware
//
#include <hip/hip_runtime.h>

// ---- problem constants (match reference) ----
#define HID   4096
#define INTER 11008
#define SEQ   2048
#define HG    32      // HID/128 groups along hidden
#define IG    86      // INTER/128 groups along inter
#define KSTEP 64
#define LDW   72      // LDS row stride in halves (144 B = 9*16 B, b128-aligned)
#define BM    128     // block tile M (seq)
#define BN    128     // block tile N (outputs)

typedef __attribute__((ext_vector_type(16))) _Float16 v16h;
typedef __attribute__((ext_vector_type(8)))  float    v8f;

union Frag16 { v16h v; unsigned int u[8]; };
union PK8    { _Float16 h[8]; uint4 q; };   // 8 halves = one 16-byte LDS store

// K-pair base inside a 32-wide K subtile for WMMA 16-bit A/B fragment VGPR v,
// per CDNA5 ISA 7.12.2 (16-bit 16x32 layout, wave32).
__device__ __forceinline__ int kpair_base(int v, int half) {
    return ((v < 4) ? (2 * v) : (16 + 2 * (v - 4))) + half * 8;
}

// dequant 8 int codes -> 8 f16
__device__ __forceinline__ void dq8(const int* c, float z, float s, PK8& p) {
    #pragma unroll
    for (int j = 0; j < 8; ++j) p.h[j] = (_Float16)(((float)c[j] - z) * s);
}

// =====================================================================
// Kernel 1: fused gate/up GEMM + SwiGLU.
// 512 thr = 16 waves (4x4), block tile 128x128, wave tile 32x32 (2x2 frags),
// K-step 64, register-pipelined staging.  blockIdx.x = M so consecutive
// blocks share the same weight stripe (L2 reuse of the int32 codes).
// =====================================================================
__global__ __launch_bounds__(512) void mlp_gateup_wmma(
    const float* __restrict__ x,
    const int*   __restrict__ gqw, const int* __restrict__ gqz,
    const float* __restrict__ gsc, const float* __restrict__ gb,
    const int*   __restrict__ uqw, const int* __restrict__ uqz,
    const float* __restrict__ usc, const float* __restrict__ ub,
    _Float16*    __restrict__ hbuf)
{
    __shared__ _Float16 sA[BM * LDW];
    __shared__ _Float16 sG[BN * LDW];
    __shared__ _Float16 sU[BN * LDW];

    const int tid  = threadIdx.x;
    const int lane = tid & 31;
    const int half = lane >> 4;
    const int lrow = lane & 15;
    const int wave = tid >> 5;      // 0..15
    const int wm   = wave & 3;      // M wave offset *32
    const int wn   = wave >> 2;     // N wave offset *32

    const int m0 = blockIdx.x * BM; // seq base (fast dim -> weight L2 reuse)
    const int n0 = blockIdx.y * BN; // inter base

    // one-shot staging ownership: 512 threads x 16 elements
    const int row = tid >> 2;       // 0..127
    const int kc  = (tid & 3) * 16; // 0,16,32,48

    const float* xa = &x  [(m0 + row) * HID + kc];
    const int*   ga = &gqw[(n0 + row) * HID + kc];
    const int*   ua = &uqw[(n0 + row) * HID + kc];

    float4 xr[4];
    int4   gc[4], uc[4];
    float  gz, gs, uz, us;

    auto load_tile = [&](int k0) {
        #pragma unroll
        for (int j = 0; j < 4; ++j) {
            xr[j] = *(const float4*)(xa + k0 + 4 * j);
            gc[j] = *(const int4*)  (ga + k0 + 4 * j);
            uc[j] = *(const int4*)  (ua + k0 + 4 * j);
        }
        const int gi = k0 >> 7;     // 64-wide tile never crosses a group
        gz = (float)gqz[(n0 + row) * HG + gi];
        gs =        gsc[(n0 + row) * HG + gi];
        uz = (float)uqz[(n0 + row) * HG + gi];
        us =        usc[(n0 + row) * HG + gi];
    };

    auto store_tile = [&]() {
        PK8 p0, p1;
        const float* xf = (const float*)xr;
        #pragma unroll
        for (int j = 0; j < 8; ++j) { p0.h[j] = (_Float16)xf[j]; p1.h[j] = (_Float16)xf[j + 8]; }
        *(uint4*)&sA[row * LDW + kc]     = p0.q;
        *(uint4*)&sA[row * LDW + kc + 8] = p1.q;

        const int* gi32 = (const int*)gc;
        dq8(gi32,     gz, gs, p0);
        dq8(gi32 + 8, gz, gs, p1);
        *(uint4*)&sG[row * LDW + kc]     = p0.q;
        *(uint4*)&sG[row * LDW + kc + 8] = p1.q;

        const int* ui32 = (const int*)uc;
        dq8(ui32,     uz, us, p0);
        dq8(ui32 + 8, uz, us, p1);
        *(uint4*)&sU[row * LDW + kc]     = p0.q;
        *(uint4*)&sU[row * LDW + kc + 8] = p1.q;
    };

    v8f accG[2][2] = {};
    v8f accU[2][2] = {};

    load_tile(0);
    for (int k0 = 0; k0 < HID; k0 += KSTEP) {
        store_tile();
        __syncthreads();
        if (k0 + KSTEP < HID) {
            __builtin_prefetch((const void*)(ga + k0 + 2 * KSTEP), 0, 1);
            __builtin_prefetch((const void*)(ua + k0 + 2 * KSTEP), 0, 1);
            load_tile(k0 + KSTEP);          // overlap next loads with WMMA
        }
        #pragma unroll
        for (int ks = 0; ks < KSTEP; ks += 32) {
            Frag16 a[2], bg[2], bu[2];
            #pragma unroll
            for (int mi = 0; mi < 2; ++mi) {
                const int ar = (wm * 32 + mi * 16 + lrow) * LDW;
                #pragma unroll
                for (int v = 0; v < 8; ++v)
                    a[mi].u[v] = *(const unsigned int*)&sA[ar + kpair_base(v, half) + ks];
            }
            #pragma unroll
            for (int ni = 0; ni < 2; ++ni) {
                const int br = (wn * 32 + ni * 16 + lrow) * LDW;
                #pragma unroll
                for (int v = 0; v < 8; ++v) {
                    const int kb = br + kpair_base(v, half) + ks;
                    bg[ni].u[v] = *(const unsigned int*)&sG[kb];
                    bu[ni].u[v] = *(const unsigned int*)&sU[kb];
                }
            }
            #pragma unroll
            for (int mi = 0; mi < 2; ++mi)
                #pragma unroll
                for (int ni = 0; ni < 2; ++ni) {
                    accG[mi][ni] = __builtin_amdgcn_wmma_f32_16x16x32_f16(
                        false, a[mi].v, false, bg[ni].v, (short)0, accG[mi][ni], false, false);
                    accU[mi][ni] = __builtin_amdgcn_wmma_f32_16x16x32_f16(
                        false, a[mi].v, false, bu[ni].v, (short)0, accU[mi][ni], false, false);
                }
        }
        __syncthreads();
    }

    // epilogue: bias + SwiGLU, store h (f16)
    #pragma unroll
    for (int ni = 0; ni < 2; ++ni) {
        const int col = n0 + wn * 32 + ni * 16 + lrow;
        const float gbv = gb[col];
        const float ubv = ub[col];
        #pragma unroll
        for (int mi = 0; mi < 2; ++mi)
            #pragma unroll
            for (int v = 0; v < 8; ++v) {
                const int r = m0 + wm * 32 + mi * 16 + v + half * 8;
                const float gv = accG[mi][ni][v] + gbv;
                const float uv = accU[mi][ni][v] + ubv;
                const float hv = (gv / (1.0f + __expf(-gv))) * uv;
                hbuf[r * INTER + col] = (_Float16)hv;
            }
    }
}

// =====================================================================
// Kernel 2: down GEMM (K = 11008).  h tile staged with CDNA5 async
// global->LDS (double-buffered, ASYNCcnt), weight dequant in VALU.
// =====================================================================
__global__ __launch_bounds__(512) void mlp_down_wmma(
    const _Float16* __restrict__ hbuf,
    const int*   __restrict__ dqw, const int* __restrict__ dqz,
    const float* __restrict__ dsc, const float* __restrict__ db,
    float*       __restrict__ y)
{
    __shared__ _Float16 sA[2][BM * LDW];   // double-buffered h tile
    __shared__ _Float16 sW[BN * LDW];

    const int tid  = threadIdx.x;
    const int lane = tid & 31;
    const int half = lane >> 4;
    const int lrow = lane & 15;
    const int wave = tid >> 5;
    const int wm   = wave & 3;
    const int wn   = wave >> 2;

    const int m0 = blockIdx.x * BM;   // seq base
    const int n0 = blockIdx.y * BN;   // hidden base

    const int row = tid >> 2;         // 0..127
    const int kc  = (tid & 3) * 16;   // halves / codes

    const _Float16* ha = &hbuf[(m0 + row) * INTER + kc];
    const int*      da = &dqw [(n0 + row) * INTER + kc];

    int4  dc[4];
    float dz, dsv;

    // CDNA5 async global->LDS: GLOBAL_LOAD_ASYNC_TO_LDS_B128 (ASYNCcnt)
    auto issue_async = [&](int k0, int buf) {
        unsigned lds0 = (unsigned)(size_t)&sA[buf][row * LDW + kc];
        unsigned lds1 = lds0 + 16;
        const _Float16* s0 = ha + k0;
        const _Float16* s1 = ha + k0 + 8;
        asm volatile("global_load_async_to_lds_b128 %0, %1, off"
                     :: "v"(lds0), "v"(s0) : "memory");
        asm volatile("global_load_async_to_lds_b128 %0, %1, off"
                     :: "v"(lds1), "v"(s1) : "memory");
    };
    auto load_w = [&](int k0) {
        #pragma unroll
        for (int j = 0; j < 4; ++j) dc[j] = *(const int4*)(da + k0 + 4 * j);
        const int gi = k0 >> 7;
        dz  = (float)dqz[(n0 + row) * IG + gi];
        dsv =        dsc[(n0 + row) * IG + gi];
    };
    auto store_w = [&]() {
        PK8 p0, p1;
        const int* c = (const int*)dc;
        dq8(c,     dz, dsv, p0);
        dq8(c + 8, dz, dsv, p1);
        *(uint4*)&sW[row * LDW + kc]     = p0.q;
        *(uint4*)&sW[row * LDW + kc + 8] = p1.q;
    };

    v8f acc[2][2] = {};

    issue_async(0, 0);
    load_w(0);
    for (int k0 = 0; k0 < INTER; k0 += KSTEP) {
        const int buf = (k0 >> 6) & 1;
        store_w();
        asm volatile("s_wait_asynccnt 0" ::: "memory");
        __syncthreads();
        if (k0 + KSTEP < INTER) {
            __builtin_prefetch((const void*)(da + k0 + 2 * KSTEP), 0, 1);
            issue_async(k0 + KSTEP, buf ^ 1);   // fill other buffer during WMMA
            load_w(k0 + KSTEP);
        }
        #pragma unroll
        for (int ks = 0; ks < KSTEP; ks += 32) {
            Frag16 a[2], b[2];
            #pragma unroll
            for (int mi = 0; mi < 2; ++mi) {
                const int ar = (wm * 32 + mi * 16 + lrow) * LDW;
                #pragma unroll
                for (int v = 0; v < 8; ++v)
                    a[mi].u[v] = *(const unsigned int*)&sA[buf][ar + kpair_base(v, half) + ks];
            }
            #pragma unroll
            for (int ni = 0; ni < 2; ++ni) {
                const int br = (wn * 32 + ni * 16 + lrow) * LDW;
                #pragma unroll
                for (int v = 0; v < 8; ++v)
                    b[ni].u[v] = *(const unsigned int*)&sW[br + kpair_base(v, half) + ks];
            }
            #pragma unroll
            for (int mi = 0; mi < 2; ++mi)
                #pragma unroll
                for (int ni = 0; ni < 2; ++ni)
                    acc[mi][ni] = __builtin_amdgcn_wmma_f32_16x16x32_f16(
                        false, a[mi].v, false, b[ni].v, (short)0, acc[mi][ni], false, false);
        }
        __syncthreads();
    }

    #pragma unroll
    for (int ni = 0; ni < 2; ++ni) {
        const int col = n0 + wn * 32 + ni * 16 + lrow;
        const float dbv = db[col];
        #pragma unroll
        for (int mi = 0; mi < 2; ++mi)
            #pragma unroll
            for (int v = 0; v < 8; ++v) {
                const int r = m0 + wm * 32 + mi * 16 + v + half * 8;
                y[r * HID + col] = acc[mi][ni][v] + dbv;
            }
    }
}

extern "C" void kernel_launch(void* const* d_in, const int* in_sizes, int n_in,
                              void* d_out, int out_size, void* d_ws, size_t ws_size,
                              hipStream_t stream) {
    const float* x   = (const float*)d_in[0];
    const int*   gqw = (const int*)  d_in[1];
    const int*   gqz = (const int*)  d_in[2];
    const float* gsc = (const float*)d_in[3];
    const float* gb  = (const float*)d_in[4];
    const int*   uqw = (const int*)  d_in[5];
    const int*   uqz = (const int*)  d_in[6];
    const float* usc = (const float*)d_in[7];
    const float* ub  = (const float*)d_in[8];
    const int*   dqw = (const int*)  d_in[9];
    const int*   dqz = (const int*)  d_in[10];
    const float* dsc = (const float*)d_in[11];
    const float* db  = (const float*)d_in[12];

    _Float16* hbuf = (_Float16*)d_ws;        // 2048*11008 f16 = 45 MB scratch
    float*    y    = (float*)d_out;

    dim3 blk(512);
    // grid.x = M tiles (fast) so blocks sharing a weight stripe run together
    mlp_gateup_wmma<<<dim3(SEQ / BM, INTER / BN), blk, 0, stream>>>(
        x, gqw, gqz, gsc, gb, uqw, uqz, usc, ub, hbuf);
    mlp_down_wmma<<<dim3(SEQ / BM, HID / BN), blk, 0, stream>>>(
        hbuf, dqw, dqz, dsc, db, y);
}
